// UNet4DMyronenko_38912403701991
// MI455X (gfx1250) — matrix operand order repaired and checked by reading, not compiled, not run on hardware
//
#include <hip/hip_runtime.h>

// ---------------------------------------------------------------------------
// 4D U-Net (Myronenko) for MI455X / gfx1250, wave32, v_wmma_f32_16x16x32_f16.
//
// Conv 3^4 as implicit GEMM with tap-major K ordering: k = tap*Cin + ci.
// All tensor dims are powers of two -> pure shift/mask addressing.
// Tap geometry + GroupNorm affine staged in LDS once per block; weights
// pre-packed into the exact 32x16 f16 B-fragment lane layout so the hot
// loop is: A-gather (branchless, LDS-table driven) + 1 B-load + prefetch
// + 1 WMMA per 32-deep K chunk. GN+ReLU fused into A-gather, residual add
// fused into the epilogue. Everything is L2-resident (192 MB).
// ---------------------------------------------------------------------------

typedef __attribute__((ext_vector_type(16))) _Float16 v16h;
typedef __attribute__((ext_vector_type(8)))  float    v8f;

struct ConvP {
    const float*    in;
    const _Float16* bpack;     // pre-packed B fragments
    const float*    bias;      // [Cout]
    const float*    gn_scale;  // [Cin]
    const float*    gn_bias;   // [Cin]
    const float*    gn_stats;  // [8*2] {mean, rstd} per group
    const float*    resid;     // [Cout, Svox] (nullable)
    float*          out;       // [Cout, Svox]
    int Cout;
    int lTi, lZi, lYi, lXi, lSin;   // log2 input dims
    int lZo, lYo, lXo, lSo;         // log2 output dims
    int lstride;                    // 0 or 1
    int nchunk, ntiles;
};

template <int LCIN, bool GN>
__global__ __launch_bounds__(256)
void conv4d_k3_wmma(ConvP p) {
    constexpr int CIN = 1 << LCIN;
    constexpr int GSH = (LCIN > 3) ? (LCIN - 3) : 0;  // channel -> group shift

    __shared__ int    s_toff[96];   // tap -> linear input offset
    __shared__ int    s_pk[96];     // tap -> packed (dt|dz<<2|dy<<4|dx<<6)
    __shared__ float2 s_gn[32];     // channel -> (scale*rstd, bias-mean*scale*rstd)

    const int tid = threadIdx.x;
    const int Zi = 1 << p.lZi, Yi = 1 << p.lYi, Xi = 1 << p.lXi, Ti = 1 << p.lTi;

    if (tid < 96) {
        const int tap = tid;
        if (tap < 81) {
            const int dt = tap / 27; int r = tap - dt * 27;
            const int dz = r / 9;  r -= dz * 9;
            const int dy = r / 3;  const int dx = r - dy * 3;
            s_toff[tid] = ((dt * Zi + dz) * Yi + dy) * Xi + dx;
            s_pk[tid]   = dt | (dz << 2) | (dy << 4) | (dx << 6);
        } else {
            s_toff[tid] = 0;
            s_pk[tid]   = 0xFF;     // dt=dz=dy=dx=3 -> always invalid
        }
    }
    if (GN && tid < CIN) {
        const int g   = tid >> GSH;
        const float mean = p.gn_stats[2 * g];
        const float rstd = p.gn_stats[2 * g + 1];
        const float sc   = p.gn_scale[tid] * rstd;
        s_gn[tid] = make_float2(sc, p.gn_bias[tid] - mean * sc);
    }
    __syncthreads();

    const int lane = tid & 31;
    const int hi   = lane >> 4;
    const int m    = lane & 15;
    const int wave = blockIdx.x * 8 + (tid >> 5);
    const int mtiles = 1 << (p.lSo - 4);            // Svox is a power of two >= 16
    if (wave >= mtiles * p.ntiles) return;
    const int mt  = wave & (mtiles - 1);
    const int ntb = wave >> (p.lSo - 4);

    // This lane's A-row output voxel (always in range: Svox % 16 == 0).
    const int vox = (mt << 4) + m;
    const int x = vox & ((1 << p.lXo) - 1);
    const int y = (vox >> p.lXo) & ((1 << p.lYo) - 1);
    const int z = (vox >> (p.lXo + p.lYo)) & ((1 << p.lZo) - 1);
    const int t = vox >> (p.lXo + p.lYo + p.lZo);

    // Input window base (pad = 1 for all 3^4 convs here).
    const int tb = (t << p.lstride) - 1;
    const int zb = (z << p.lstride) - 1;
    const int yb = (y << p.lstride) - 1;
    const int xb = (x << p.lstride) - 1;
    const int base = ((tb * Zi + zb) * Yi + yb) * Xi + xb;

    // Per-dimension 3-bit validity masks (bit d: in-range at offset d).
    unsigned mT = 0, mZ = 0, mY = 0, mX = 0;
    #pragma unroll
    for (int d = 0; d < 3; ++d) {
        mT |= ((unsigned)(tb + d) < (unsigned)Ti) << d;
        mZ |= ((unsigned)(zb + d) < (unsigned)Zi) << d;
        mY |= ((unsigned)(yb + d) < (unsigned)Yi) << d;
        mX |= ((unsigned)(xb + d) < (unsigned)Xi) << d;
    }

    const _Float16* bp = p.bpack + (size_t)(ntb * p.nchunk) * 512 + lane * 16;

    v8f acc = {};
    for (int kc = 0; kc < p.nchunk; ++kc) {
        // ---- B fragment: direct aligned load of pre-packed lane data ----
        const v16h B = *(const v16h*)(bp + (size_t)kc * 512);
        if (kc + 1 < p.nchunk)
            __builtin_prefetch((const void*)(bp + (size_t)(kc + 1) * 512), 0, 3);

        // ---- A fragment: branchless LDS-table-driven gather ----
        const int kb = kc << 5;
        v16h A;
        #pragma unroll
        for (int h = 0; h < 16; ++h) {
            const int klocal = (hi << 4) + h;
            const int ci  = klocal & (CIN - 1);
            const int tap = (kb + klocal) >> LCIN;
            const int toff = s_toff[tap];
            const int pk   = s_pk[tap];
            const unsigned ok = (mT >> (pk & 3)) & (mZ >> ((pk >> 2) & 3)) &
                                (mY >> ((pk >> 4) & 3)) & (mX >> ((pk >> 6) & 3)) & 1u;
            const int addr = (ci << p.lSin) + base + toff;
            float v = p.in[ok ? addr : 0];
            if (GN) {
                const float2 gg = s_gn[ci];
                v = fmaf(v, gg.x, gg.y);
                v = fmaxf(v, 0.0f);
            }
            A[h] = ok ? (_Float16)v : (_Float16)0.0f;
        }

        acc = __builtin_amdgcn_wmma_f32_16x16x32_f16(
                  false, A, false, B, (short)0, acc, false, false);
    }

    // ---- Epilogue: D f32 layout (VGPR r -> M = r + hi*8, N = lane&15) ----
    const int n = (ntb << 4) + m;
    if (n < p.Cout) {
        const int Svox = 1 << p.lSo;
        const float bn = p.bias[n];
        const long ibase = (long)n * Svox + (mt << 4) + (hi << 3);
        const float* rp = p.resid;
        if (rp) {                       // uniform branch, hoisted out of loop
            #pragma unroll
            for (int r = 0; r < 8; ++r)
                p.out[ibase + r] = acc[r] + bn + rp[ibase + r];
        } else {
            #pragma unroll
            for (int r = 0; r < 8; ++r)
                p.out[ibase + r] = acc[r] + bn;
        }
    }
}

// ---------------------------------------------------------------------------
// Weight pre-pack: f32 [Cout,Cin,81] -> f16 B fragments in the conv kernel's (A-matching) lane
// layout (K = (lane>>4)*16 + h, N = lane&15), tap-major K = tap*Cin + ci.
// ---------------------------------------------------------------------------
__global__ __launch_bounds__(256)
void pack_weights_kernel(const float* w, _Float16* bp,
                         int Cin, int lcin, int Cout, int nchunk, int ntiles) {
    const int i = blockIdx.x * blockDim.x + threadIdx.x;
    const int total = ntiles * nchunk * 512;
    if (i >= total) return;
    const int h    = i & 15;
    const int lane = (i >> 4) & 31;
    const int c    = i >> 9;
    const int kc   = c % nchunk;
    const int nt   = c / nchunk;
    const int n    = nt * 16 + (lane & 15);
    const int k    = kc * 32 + ((lane >> 4) << 4) + h;
    const int ci   = k & (Cin - 1);
    const int tap  = k >> lcin;
    float v = 0.0f;
    if (tap < 81 && n < Cout) v = w[(n * Cin + ci) * 81 + tap];
    bp[i] = (_Float16)v;
}

// ---------------------------------------------------------------------------
// GroupNorm statistics: one block per group (GROUPS = 8); a group's channels
// are a contiguous [cpg*S] slab in channel-major layout.
// ---------------------------------------------------------------------------
__global__ __launch_bounds__(256)
void gn_stats_kernel(const float* in, float* stats, int C, int S) {
    const int g   = blockIdx.x;
    const int cpg = C >> 3;
    const long n  = (long)cpg * S;
    const float* base = in + (long)g * n;

    float s = 0.0f, ss = 0.0f;
    for (long i = threadIdx.x; i < n; i += blockDim.x) {
        const float v = base[i];
        s += v; ss += v * v;
    }
    __shared__ float red[512];
    red[threadIdx.x]       = s;
    red[256 + threadIdx.x] = ss;
    __syncthreads();
    for (int off = 128; off > 0; off >>= 1) {
        if ((int)threadIdx.x < off) {
            red[threadIdx.x]       += red[threadIdx.x + off];
            red[256 + threadIdx.x] += red[256 + threadIdx.x + off];
        }
        __syncthreads();
    }
    if (threadIdx.x == 0) {
        const float inv  = 1.0f / (float)n;
        const float mean = red[0] * inv;
        const float var  = red[256] * inv - mean * mean;
        stats[2 * g]     = mean;
        stats[2 * g + 1] = rsqrtf(var + 1e-5f);
    }
}

// ---------------------------------------------------------------------------
// Fused: out = bypass + upsample2(conv1x1(hlo)).  Pads are all zero for these
// shapes; nearest upsample = coordinate >> 1.
// ---------------------------------------------------------------------------
__global__ __launch_bounds__(256)
void up_fuse_kernel(const float* bypass, const float* hlo,
                    const float* w, const float* b, float* out,
                    int Cout, int Cin, int T, int Z, int Y, int X) {
    const long S = (long)T * Z * Y * X;
    const long i = (long)blockIdx.x * blockDim.x + threadIdx.x;
    if (i >= S) return;
    long r = i;
    const int x = r % X; r /= X;
    const int y = r % Y; r /= Y;
    const int z = r % Z; const int t = (int)(r / Z);
    const int Zl = Z >> 1, Yl = Y >> 1, Xl = X >> 1, Tl = T >> 1;
    const long Slo = (long)Tl * Zl * Yl * Xl;
    const long vlo = (((long)(t >> 1) * Zl + (z >> 1)) * Yl + (y >> 1)) * Xl + (x >> 1);
    for (int nn = 0; nn < Cout; ++nn) {
        float acc = b[nn];
        for (int ci = 0; ci < Cin; ++ci)
            acc += w[nn * Cin + ci] * hlo[ci * Slo + vlo];
        out[nn * S + i] = bypass[nn * S + i] + acc;
    }
}

// ---------------------------------------------------------------------------
// Fused end: 1x1 conv (8 -> 2) + softmax over the 2 channels.
// ---------------------------------------------------------------------------
__global__ __launch_bounds__(256)
void end_softmax_kernel(const float* in, const float* w, const float* b,
                        float* out, long S) {
    const long i = (long)blockIdx.x * blockDim.x + threadIdx.x;
    if (i >= S) return;
    float l0 = b[0], l1 = b[1];
    #pragma unroll
    for (int ci = 0; ci < 8; ++ci) {
        const float v = in[ci * S + i];
        l0 += w[ci] * v;
        l1 += w[8 + ci] * v;
    }
    const float mx = fmaxf(l0, l1);
    const float e0 = __expf(l0 - mx), e1 = __expf(l1 - mx);
    const float inv = 1.0f / (e0 + e1);
    out[i]     = e0 * inv;
    out[S + i] = e1 * inv;
}

// ---------------------------------------------------------------------------
// Host-side orchestration
// ---------------------------------------------------------------------------
static inline int cdiv(int a, int b) { return (a + b - 1) / b; }

static void launch_conv(hipStream_t s, const float* in, const _Float16* bpack,
                        const float* bias, const float* gs, const float* gb,
                        const float* stats, const float* resid, float* out,
                        int Cin, int Cout, int lT, int lZ, int lY, int lX,
                        int lstride, int gn) {
    ConvP p;
    p.in = in; p.bpack = bpack; p.bias = bias;
    p.gn_scale = gs; p.gn_bias = gb; p.gn_stats = stats;
    p.resid = resid; p.out = out; p.Cout = Cout;
    p.lTi = lT; p.lZi = lZ; p.lYi = lY; p.lXi = lX;
    p.lSin = lT + lZ + lY + lX;
    p.lZo = lZ - lstride; p.lYo = lY - lstride; p.lXo = lX - lstride;
    p.lSo = p.lSin - 4 * lstride;
    p.lstride = lstride;
    p.nchunk = cdiv(Cin * 81, 32);
    p.ntiles = cdiv(Cout, 16);
    const int waves  = (1 << (p.lSo - 4)) * p.ntiles;
    const int blocks = cdiv(waves, 8);
    if (gn) {
        if (Cin == 8)       conv4d_k3_wmma<3, true><<<blocks, 256, 0, s>>>(p);
        else if (Cin == 16) conv4d_k3_wmma<4, true><<<blocks, 256, 0, s>>>(p);
        else                conv4d_k3_wmma<5, true><<<blocks, 256, 0, s>>>(p);
    } else {
        if (Cin == 4)       conv4d_k3_wmma<2, false><<<blocks, 256, 0, s>>>(p);
        else if (Cin == 8)  conv4d_k3_wmma<3, false><<<blocks, 256, 0, s>>>(p);
        else                conv4d_k3_wmma<4, false><<<blocks, 256, 0, s>>>(p);
    }
}

// GroupNorm->ReLU->conv, GroupNorm->ReLU->conv(+residual)
static void run_block(hipStream_t s, const float* in, float* tmp, float* out,
                      const float* g1s, const float* g1b,
                      const _Float16* c1p, const float* c1b,
                      const float* g2s, const float* g2b,
                      const _Float16* c2p, const float* c2b,
                      float* stA, float* stB,
                      int C, int lT, int lZ, int lY, int lX) {
    const int S = 1 << (lT + lZ + lY + lX);
    gn_stats_kernel<<<8, 256, 0, s>>>(in, stA, C, S);
    launch_conv(s, in, c1p, c1b, g1s, g1b, stA, nullptr, tmp, C, C, lT, lZ, lY, lX, 0, 1);
    gn_stats_kernel<<<8, 256, 0, s>>>(tmp, stB, C, S);
    launch_conv(s, tmp, c2p, c2b, g2s, g2b, stB, in, out, C, C, lT, lZ, lY, lX, 0, 1);
}

// Input ordering (setup_inputs dict order, nested dicts flattened in order):
enum {
    IN_X = 0,
    IN_INIT_W, IN_INIT_B,
    EB0_G1S, EB0_G1B, EB0_G2S, EB0_G2B, EB0_C1W, EB0_C1B, EB0_C2W, EB0_C2B,
    DN1_W, DN1_B,
    EB1_G1S, EB1_G1B, EB1_G2S, EB1_G2B, EB1_C1W, EB1_C1B, EB1_C2W, EB1_C2B,
    DN2_W, DN2_B,
    EB2_G1S, EB2_G1B, EB2_G2S, EB2_G2B, EB2_C1W, EB2_C1B, EB2_C2W, EB2_C2B,
    UP1_W, UP1_B,
    DB1_G1S, DB1_G1B, DB1_G2S, DB1_G2B, DB1_C1W, DB1_C1B, DB1_C2W, DB1_C2B,
    UP0_W, UP0_B,
    DB0_G1S, DB0_G1B, DB0_G2S, DB0_G2B, DB0_C1W, DB0_C1B, DB0_C2W, DB0_C2B,
    END_W, END_B
};

extern "C" void kernel_launch(void* const* d_in, const int* in_sizes, int n_in,
                              void* d_out, int out_size, void* d_ws, size_t ws_size,
                              hipStream_t stream) {
    const float* F[64];
    for (int i = 0; i < n_in && i < 64; ++i) F[i] = (const float*)d_in[i];

    const int S0 = 16 * 32 * 32 * 32;  // 524288
    const int S1 = 8 * 16 * 16 * 16;   // 32768
    const int S2 = 4 * 8 * 8 * 8;      // 2048

    // ---- Workspace carve-up: activations (f32), then weight packs (f16) ----
    float* p   = (float*)d_ws;
    float* stA = p;  p += 16;
    float* stB = p;  p += 16;
    float* ic  = p;  p += 8 * S0;   // init conv out; later reused as du0
    float* tf  = p;  p += 8 * S0;   // full-res block temp
    float* eb0 = p;  p += 8 * S0;   // skip (kept)
    float* dbf = p;  p += 8 * S0;   // db0
    float* ed1 = p;  p += 16 * S1;
    float* e1x = p;  p += 16 * S1;  // eb10
    float* e1y = p;  p += 16 * S1;  // eb11 (skip, kept)
    float* du1 = p;  p += 16 * S1;
    float* db1 = p;  p += 16 * S1;
    float* t1  = p;  p += 16 * S1;  // level-1 block temp
    float* h2a = p;  p += 32 * S2;
    float* h2b = p;  p += 32 * S2;
    float* t2  = p;  p += 32 * S2;  // level-2 block temp
    _Float16* hp = (_Float16*)p;
    (void)ws_size;

    // ---- Pre-pack all 3^4 conv weights into f16 B-fragment layout ----
    auto packw = [&](const float* w, int Cin, int lcin, int Cout) -> _Float16* {
        const int nchunk = cdiv(Cin * 81, 32);
        const int ntiles = cdiv(Cout, 16);
        const int total  = ntiles * nchunk * 512;
        _Float16* dst = hp; hp += total;
        pack_weights_kernel<<<cdiv(total, 256), 256, 0, stream>>>(
            w, dst, Cin, lcin, Cout, nchunk, ntiles);
        return dst;
    };
    _Float16* P_init  = packw(F[IN_INIT_W], 4, 2, 8);
    _Float16* P_e0c1  = packw(F[EB0_C1W],  8, 3, 8);
    _Float16* P_e0c2  = packw(F[EB0_C2W],  8, 3, 8);
    _Float16* P_dn1   = packw(F[DN1_W],    8, 3, 16);
    _Float16* P_e1c1  = packw(F[EB1_C1W], 16, 4, 16);
    _Float16* P_e1c2  = packw(F[EB1_C2W], 16, 4, 16);
    _Float16* P_dn2   = packw(F[DN2_W],   16, 4, 32);
    _Float16* P_e2c1  = packw(F[EB2_C1W], 32, 5, 32);
    _Float16* P_e2c2  = packw(F[EB2_C2W], 32, 5, 32);
    _Float16* P_d1c1  = packw(F[DB1_C1W], 16, 4, 16);
    _Float16* P_d1c2  = packw(F[DB1_C2W], 16, 4, 16);
    _Float16* P_d0c1  = packw(F[DB0_C1W],  8, 3, 8);
    _Float16* P_d0c2  = packw(F[DB0_C2W],  8, 3, 8);

    // log2 dims: full (4,5,5,5), L1 (3,4,4,4), L2 (2,3,3,3)
    // 1. init conv: 4 -> 8, full res, no GN
    launch_conv(stream, F[IN_X], P_init, F[IN_INIT_B],
                nullptr, nullptr, nullptr, nullptr, ic,
                4, 8, 4, 5, 5, 5, 0, 0);

    // 2. eb0 = conv_block(ic) + ic
    run_block(stream, ic, tf, eb0,
              F[EB0_G1S], F[EB0_G1B], P_e0c1, F[EB0_C1B],
              F[EB0_G2S], F[EB0_G2B], P_e0c2, F[EB0_C2B],
              stA, stB, 8, 4, 5, 5, 5);

    // 3. ed1 = down1(eb0), stride 2: 8 -> 16
    launch_conv(stream, eb0, P_dn1, F[DN1_B],
                nullptr, nullptr, nullptr, nullptr, ed1,
                8, 16, 4, 5, 5, 5, 1, 0);

    // 4-5. eb1 applied twice (shared weights)
    run_block(stream, ed1, t1, e1x,
              F[EB1_G1S], F[EB1_G1B], P_e1c1, F[EB1_C1B],
              F[EB1_G2S], F[EB1_G2B], P_e1c2, F[EB1_C2B],
              stA, stB, 16, 3, 4, 4, 4);
    run_block(stream, e1x, t1, e1y,
              F[EB1_G1S], F[EB1_G1B], P_e1c1, F[EB1_C1B],
              F[EB1_G2S], F[EB1_G2B], P_e1c2, F[EB1_C2B],
              stA, stB, 16, 3, 4, 4, 4);

    // 6. h = down2(eb11), stride 2: 16 -> 32
    launch_conv(stream, e1y, P_dn2, F[DN2_B],
                nullptr, nullptr, nullptr, nullptr, h2a,
                16, 32, 3, 4, 4, 4, 1, 0);

    // 7. eb2 applied 4x (shared weights), ping-pong
    float* cur = h2a; float* alt = h2b;
    for (int i = 0; i < 4; ++i) {
        run_block(stream, cur, t2, alt,
                  F[EB2_G1S], F[EB2_G1B], P_e2c1, F[EB2_C1B],
                  F[EB2_G2S], F[EB2_G2B], P_e2c2, F[EB2_C2B],
                  stA, stB, 32, 2, 3, 3, 3);
        float* sw = cur; cur = alt; alt = sw;
    }

    // 8. du1 = eb11 + upsample2(up1(h))
    up_fuse_kernel<<<cdiv(S1, 256), 256, 0, stream>>>(
        e1y, cur, F[UP1_W], F[UP1_B], du1, 16, 32, 8, 16, 16, 16);

    // 9. db1 = conv_block(du1) + du1
    run_block(stream, du1, t1, db1,
              F[DB1_G1S], F[DB1_G1B], P_d1c1, F[DB1_C1B],
              F[DB1_G2S], F[DB1_G2B], P_d1c2, F[DB1_C2B],
              stA, stB, 16, 3, 4, 4, 4);

    // 10. du0 = eb0 + upsample2(up0(db1))  (reuse ic)
    up_fuse_kernel<<<cdiv(S0, 256), 256, 0, stream>>>(
        eb0, db1, F[UP0_W], F[UP0_B], ic, 8, 16, 16, 32, 32, 32);

    // 11. db0 = conv_block(du0) + du0
    run_block(stream, ic, tf, dbf,
              F[DB0_G1S], F[DB0_G1B], P_d0c1, F[DB0_C1B],
              F[DB0_G2S], F[DB0_G2B], P_d0c2, F[DB0_C2B],
              stA, stB, 8, 4, 5, 5, 5);

    // 12. logits = end(db0); softmax over 2 channels
    end_softmax_kernel<<<cdiv(S0, 256), 256, 0, stream>>>(
        dbf, F[END_W], F[END_B], (float*)d_out, (long)S0);
    (void)out_size; (void)in_sizes;
}
